// TranformerBlock_32495722562184
// MI455X (gfx1250) — matrix operand order repaired and checked by reading, compiled, not run on hardware
//
#include <hip/hip_runtime.h>
#include <stdint.h>

#define Bb   2
#define Tt   2048
#define Hh   768
#define NHh  12
#define HDd  64
#define DFFf 3072
#define BT   (Bb*Tt)

typedef uint16_t u16;
typedef __attribute__((ext_vector_type(16))) __bf16 v16bf;
typedef __attribute__((ext_vector_type(8)))  float  v8f;

__device__ __forceinline__ u16 f2bf(float f) {
    uint32_t u = __float_as_uint(f);
    u += 0x7FFFu + ((u >> 16) & 1u);   // round-to-nearest-even
    return (u16)(u >> 16);
}

__device__ __forceinline__ v8f wmma_bf16(v16bf a, v16bf b, v8f c) {
    return __builtin_amdgcn_wmma_f32_16x16x32_bf16(false, a, false, b, (short)0, c, false, false);
}

// async global -> LDS copy of 16 bytes per lane (CDNA5 ASYNCcnt path)
__device__ __forceinline__ void async_copy_b128(uint32_t lds_byte_addr, const void* gaddr) {
    uint64_t ga = (uint64_t)(uintptr_t)gaddr;
    asm volatile("global_load_async_to_lds_b128 %0, %1, off"
                 :: "v"(lds_byte_addr), "v"(ga) : "memory");
}

// ---------------------------------------------------------------- add: dst = a + b
__global__ __launch_bounds__(256) void add2_k(float* dst, const float* a, const float* b) {
    size_t i = ((size_t)blockIdx.x * 256 + threadIdx.x) * 4;
    float4 av = *reinterpret_cast<const float4*>(a + i);
    float4 bv = *reinterpret_cast<const float4*>(b + i);
    av.x += bv.x; av.y += bv.y; av.z += bv.z; av.w += bv.w;
    *reinterpret_cast<float4*>(dst + i) = av;
}

// ---------------------------------------------------------------- weight fp32[K,N] -> bf16[N,K]
__global__ __launch_bounds__(256) void convw_k(const float* __restrict__ W,
                                               u16* __restrict__ Wt, int K, int N) {
    size_t i = (size_t)blockIdx.x * 256 + threadIdx.x;
    if (i >= (size_t)K * N) return;
    size_t k = i / N, n = i % N;
    Wt[n * (size_t)K + k] = f2bf(W[i]);
}

// ---------------------------------------------------------------- LayerNorm, one row per block
__global__ __launch_bounds__(256) void ln_bf16_k(const float* __restrict__ x,
        const float* __restrict__ sc, const float* __restrict__ bi, u16* __restrict__ out) {
    __shared__ float rs[256], rq[256];
    const int tid = threadIdx.x;
    const size_t row = blockIdx.x;
    const float* xr = x + row * Hh;
    float v0 = xr[tid], v1 = xr[tid + 256], v2 = xr[tid + 512];
    rs[tid] = v0 + v1 + v2;
    rq[tid] = v0*v0 + v1*v1 + v2*v2;
    __syncthreads();
    for (int o = 128; o > 0; o >>= 1) {
        if (tid < o) { rs[tid] += rs[tid + o]; rq[tid] += rq[tid + o]; }
        __syncthreads();
    }
    float mean = rs[0] * (1.0f / 768.0f);
    float var  = rq[0] * (1.0f / 768.0f) - mean * mean;
    float rinv = rsqrtf(var + 1e-5f);
    u16* orow = out + row * Hh;
    orow[tid]       = f2bf((v0 - mean) * rinv * sc[tid]       + bi[tid]);
    orow[tid + 256] = f2bf((v1 - mean) * rinv * sc[tid + 256] + bi[tid + 256]);
    orow[tid + 512] = f2bf((v2 - mean) * rinv * sc[tid + 512] + bi[tid + 512]);
}

__global__ __launch_bounds__(256) void ln_f32_k(const float* __restrict__ x,
        const float* __restrict__ sc, const float* __restrict__ bi, float* __restrict__ out) {
    __shared__ float rs[256], rq[256];
    const int tid = threadIdx.x;
    const size_t row = blockIdx.x;
    const float* xr = x + row * Hh;
    float v0 = xr[tid], v1 = xr[tid + 256], v2 = xr[tid + 512];
    rs[tid] = v0 + v1 + v2;
    rq[tid] = v0*v0 + v1*v1 + v2*v2;
    __syncthreads();
    for (int o = 128; o > 0; o >>= 1) {
        if (tid < o) { rs[tid] += rs[tid + o]; rq[tid] += rq[tid + o]; }
        __syncthreads();
    }
    float mean = rs[0] * (1.0f / 768.0f);
    float var  = rq[0] * (1.0f / 768.0f) - mean * mean;
    float rinv = rsqrtf(var + 1e-5f);
    float* orow = out + row * Hh;
    orow[tid]       = (v0 - mean) * rinv * sc[tid]       + bi[tid];
    orow[tid + 256] = (v1 - mean) * rinv * sc[tid + 256] + bi[tid + 256];
    orow[tid + 512] = (v2 - mean) * rinv * sc[tid + 512] + bi[tid + 512];
}

// ---------------------------------------------------------------- one K-chunk of the block GEMM
__device__ __forceinline__ void gemm_step(const u16* __restrict__ arow0,
        const u16* __restrict__ arow1, int kk, const u16* __restrict__ bbuf,
        int m16, int hf, v8f acc[2][4]) {
    v16bf af0, af1;
    uint4* ap = reinterpret_cast<uint4*>(&af0);
    ap[0] = *reinterpret_cast<const uint4*>(arow0 + kk);
    ap[1] = *reinterpret_cast<const uint4*>(arow0 + kk + 16);
    uint4* aq = reinterpret_cast<uint4*>(&af1);
    aq[0] = *reinterpret_cast<const uint4*>(arow1 + kk);
    aq[1] = *reinterpret_cast<const uint4*>(arow1 + kk + 16);
    #pragma unroll
    for (int j = 0; j < 4; ++j) {
        const u16* bp16 = bbuf + (j*16 + m16) * 32 + hf * 8;
        v16bf bfv;
        uint4* bp = reinterpret_cast<uint4*>(&bfv);
        bp[0] = *reinterpret_cast<const uint4*>(bp16);
        bp[1] = *reinterpret_cast<const uint4*>(bp16 + 16);
        acc[0][j] = wmma_bf16(af0, bfv, acc[0][j]);
        acc[1][j] = wmma_bf16(af1, bfv, acc[1][j]);
    }
}

// ---------------------------------------------------------------- GEMM: C[M=4096,N] = A[M,K]bf16 * Wt[N,K]bf16 + bias
// Block = 256 rows x 64 cols; 8 waves, each wave 32 rows (2 M-tiles) x 64 cols (4 N-tiles).
// B tile (64x32 bf16 = 4KB) staged via async global->LDS, double buffered, last step peeled.
// mode 1: bf16 out in [b,nh,t,d]   (Q,K)
// mode 2: bf16 out in [b,nh,d,t]   (V transposed)
// mode 3: f32  out[row*N+col] += v (residual add)
// mode 4: bf16 out = gelu(v)       (FFN intermediate)
__global__ __launch_bounds__(256) void gemm_bf16_k(const u16* __restrict__ A,
        const u16* __restrict__ Wt, const float* __restrict__ bias,
        void* __restrict__ outp, int N, int K, int mode) {
    __shared__ __align__(16) u16 sB[2][64 * 32];
    const int tid  = threadIdx.x;
    const int lane = tid & 31;
    const int wave = tid >> 5;
    const int m16  = lane & 15;
    const int hf   = lane >> 4;
    const int row0 = blockIdx.x * 256 + wave * 32;
    const int ncol0 = blockIdx.y * 64;

    // cooperative async-copy mapping: thread -> (B row, 16B chunk)
    const int bn    = tid >> 2;   // 0..63
    const int bpart = tid & 3;    // 0..3
    const u16* gsrcBase = Wt + (size_t)(ncol0 + bn) * K + bpart * 8;
    const uint32_t ldsBase = (uint32_t)(uintptr_t)(&sB[0][0]);
    const uint32_t ldsDst0 = ldsBase + (uint32_t)(bn * 64 + bpart * 16);

    v8f zero = {0.f,0.f,0.f,0.f,0.f,0.f,0.f,0.f};
    v8f acc[2][4] = {{zero, zero, zero, zero}, {zero, zero, zero, zero}};

    const u16* arow0 = A + (size_t)(row0 + m16) * K + hf * 8;
    const u16* arow1 = A + (size_t)(row0 + 16 + m16) * K + hf * 8;

    // prologue: stage first K-chunk
    async_copy_b128(ldsDst0, gsrcBase);

    int buf = 0;
    int kk = 0;
    // steady state: unconditional prefetch of next chunk, compute current
    for (; kk < K - 32; kk += 32) {
        async_copy_b128(ldsDst0 + (uint32_t)((buf ^ 1) * 4096), gsrcBase + kk + 32);
        asm volatile("s_wait_asynccnt 0x1" ::: "memory");
        __syncthreads();
        gemm_step(arow0, arow1, kk, &sB[buf][0], m16, hf, acc);
        __syncthreads();   // all waves done reading buf before next overwrite
        buf ^= 1;
    }
    // peeled final chunk
    asm volatile("s_wait_asynccnt 0x0" ::: "memory");
    __syncthreads();
    gemm_step(arow0, arow1, kk, &sB[buf][0], m16, hf, acc);

    #pragma unroll
    for (int mt = 0; mt < 2; ++mt) {
        #pragma unroll
        for (int j = 0; j < 4; ++j) {
            const int col = ncol0 + j*16 + m16;
            const float bvv = bias[col];
            #pragma unroll
            for (int r = 0; r < 8; ++r) {
                const int row = row0 + mt*16 + r + hf * 8;
                float v = acc[mt][j][r] + bvv;
                if (mode == 1) {
                    int b = row >> 11, t = row & 2047;
                    int nh = col >> 6, d = col & 63;
                    ((u16*)outp)[(((size_t)b * NHh + nh) * Tt + t) * HDd + d] = f2bf(v);
                } else if (mode == 2) {
                    int b = row >> 11, t = row & 2047;
                    int nh = col >> 6, d = col & 63;
                    ((u16*)outp)[(((size_t)b * NHh + nh) * HDd + d) * Tt + t] = f2bf(v);
                } else if (mode == 3) {
                    ((float*)outp)[(size_t)row * N + col] += v;
                } else {
                    float inner = 0.7978845608028654f * (v + 0.044715f * v * v * v);
                    float e  = __expf(2.0f * inner);
                    float th = 1.0f - 2.0f / (e + 1.0f);
                    ((u16*)outp)[(size_t)row * N + col] = f2bf(0.5f * v * (1.0f + th));
                }
            }
        }
    }
}

// ---------------------------------------------------------------- flash attention, 1 wave = 16 query rows
__global__ __launch_bounds__(128) void attn_k(const u16* __restrict__ q,
        const u16* __restrict__ kmat, const u16* __restrict__ vT,
        const float* __restrict__ mask, u16* __restrict__ av) {
    __shared__ __align__(16) u16 lds_p[4][16 * 32];
    const int lane = threadIdx.x & 31;
    const int wave = threadIdx.x >> 5;
    const int m16  = lane & 15;
    const int hf   = lane >> 4;
    const int bh   = blockIdx.x;
    const int b = bh / NHh, nh = bh % NHh;
    const int t0 = blockIdx.y * 64 + wave * 16;
    u16* pbuf = lds_p[wave];

    const u16* qbase = q + ((size_t)bh * Tt + t0) * HDd;
    v16bf qa[2];
    #pragma unroll
    for (int c = 0; c < 2; ++c) {
        uint4* qp = reinterpret_cast<uint4*>(&qa[c]);
        qp[0] = *reinterpret_cast<const uint4*>(qbase + m16 * HDd + c*32 + hf*8);
        qp[1] = *reinterpret_cast<const uint4*>(qbase + m16 * HDd + c*32 + 16 + hf*8);
    }

    float rowM[8], rowS[8];
    #pragma unroll
    for (int r = 0; r < 8; ++r) { rowM[r] = -3.0e38f; rowS[r] = 0.f; }
    v8f zero = {0.f,0.f,0.f,0.f,0.f,0.f,0.f,0.f};
    v8f O[4] = {zero, zero, zero, zero};

    const u16* kbase = kmat + (size_t)bh * Tt * HDd;
    const u16* vbase = vT   + (size_t)bh * HDd * Tt;
    const float* mrow = mask + (size_t)b * Tt;

    for (int tp = 0; tp < Tt; tp += 32) {
        v8f S[2] = {zero, zero};
        #pragma unroll
        for (int j = 0; j < 2; ++j) {
            const u16* kp0 = kbase + (size_t)(tp + j*16 + m16) * HDd + hf * 8;
            #pragma unroll
            for (int c = 0; c < 2; ++c) {
                v16bf kb;
                uint4* kp = reinterpret_cast<uint4*>(&kb);
                kp[0] = *reinterpret_cast<const uint4*>(kp0 + c*32);
                kp[1] = *reinterpret_cast<const uint4*>(kp0 + c*32 + 16);
                S[j] = wmma_bf16(qa[c], kb, S[j]);
            }
        }
        #pragma unroll
        for (int j = 0; j < 2; ++j) {
            float mv  = mrow[tp + j*16 + m16];
            float ofs = -1.0e10f * (1.0f - mv);
            #pragma unroll
            for (int r = 0; r < 8; ++r)
                S[j][r] = S[j][r] * 0.125f * mv + ofs;
        }
        float corr[8];
        #pragma unroll
        for (int r = 0; r < 8; ++r) {
            float t = fmaxf(S[0][r], S[1][r]);
            #pragma unroll
            for (int o = 1; o < 16; o <<= 1)
                t = fmaxf(t, __shfl_xor(t, o, 32));
            float nm = fmaxf(rowM[r], t);
            corr[r] = __expf(rowM[r] - nm);
            rowM[r] = nm;
        }
        #pragma unroll
        for (int j = 0; j < 2; ++j)
            #pragma unroll
            for (int r = 0; r < 8; ++r)
                S[j][r] = __expf(S[j][r] - rowM[r]);
        #pragma unroll
        for (int r = 0; r < 8; ++r) {
            float ps = S[0][r] + S[1][r];
            #pragma unroll
            for (int o = 1; o < 16; o <<= 1)
                ps += __shfl_xor(ps, o, 32);
            rowS[r] = rowS[r] * corr[r] + ps;
            #pragma unroll
            for (int dt = 0; dt < 4; ++dt)
                O[dt][r] *= corr[r];
        }
        // C-layout -> A-layout via per-wave LDS (DS ops are in-order within a wave)
        #pragma unroll
        for (int j = 0; j < 2; ++j)
            #pragma unroll
            for (int r = 0; r < 8; ++r)
                pbuf[(r + hf*8) * 32 + j*16 + m16] = f2bf(S[j][r]);
        v16bf pa;
        uint4* pp = reinterpret_cast<uint4*>(&pa);
        pp[0] = *reinterpret_cast<const uint4*>(pbuf + m16*32 + hf*8);
        pp[1] = *reinterpret_cast<const uint4*>(pbuf + m16*32 + 16 + hf*8);
        #pragma unroll
        for (int dt = 0; dt < 4; ++dt) {
            const u16* vp = vbase + (size_t)(dt*16 + m16) * Tt + tp + hf * 8;
            v16bf vb;
            uint4* vv = reinterpret_cast<uint4*>(&vb);
            vv[0] = *reinterpret_cast<const uint4*>(vp);
            vv[1] = *reinterpret_cast<const uint4*>(vp + 16);
            O[dt] = wmma_bf16(pa, vb, O[dt]);
        }
    }

    #pragma unroll
    for (int r = 0; r < 8; ++r) {
        float inv = 1.0f / rowS[r];
        const size_t rowbase = ((size_t)b * Tt + (size_t)(t0 + r + hf*8)) * Hh + nh * HDd;
        #pragma unroll
        for (int dt = 0; dt < 4; ++dt)
            av[rowbase + dt*16 + m16] = f2bf(O[dt][r] * inv);
    }
}

// ----------------------------------------------------------------
extern "C" void kernel_launch(void* const* d_in, const int* in_sizes, int n_in,
                              void* d_out, int out_size, void* d_ws, size_t ws_size,
                              hipStream_t stream) {
    (void)in_sizes; (void)n_in; (void)out_size; (void)ws_size;
    const float* in_h  = (const float*)d_in[0];
    const float* embs  = (const float*)d_in[1];
    const float* mask  = (const float*)d_in[2];
    const float* ln1_s = (const float*)d_in[3];
    const float* ln1_b = (const float*)d_in[4];
    const float* wq    = (const float*)d_in[5];
    const float* bq    = (const float*)d_in[6];
    const float* wk    = (const float*)d_in[7];
    const float* bk    = (const float*)d_in[8];
    const float* wv    = (const float*)d_in[9];
    const float* bv    = (const float*)d_in[10];
    const float* wo    = (const float*)d_in[11];
    const float* bo    = (const float*)d_in[12];
    const float* ln2_s = (const float*)d_in[13];
    const float* ln2_b = (const float*)d_in[14];
    const float* w1    = (const float*)d_in[15];
    const float* b1    = (const float*)d_in[16];
    const float* w2    = (const float*)d_in[17];
    const float* b2    = (const float*)d_in[18];
    const float* lnf_s = (const float*)d_in[19];
    const float* lnf_b = (const float*)d_in[20];

    const size_t nBTH = (size_t)BT * Hh;          // 3,145,728
    float* h_cur = (float*)d_ws;
    u16* p   = (u16*)(h_cur + nBTH);
    u16* hn  = p; p += nBTH;
    u16* qb  = p; p += nBTH;
    u16* kb  = p; p += nBTH;
    u16* vtb = p; p += nBTH;
    u16* avb = p; p += nBTH;
    u16* ffn = p; p += (size_t)BT * DFFf;
    u16 *wtq[2], *wtk[2], *wtv[2], *wto[2], *wt1[2], *wt2[2];
    for (int l = 0; l < 2; ++l) {
        wtq[l] = p; p += Hh * Hh;
        wtk[l] = p; p += Hh * Hh;
        wtv[l] = p; p += Hh * Hh;
        wto[l] = p; p += Hh * Hh;
        wt1[l] = p; p += (size_t)Hh * DFFf;
        wt2[l] = p; p += (size_t)Hh * DFFf;
    }

    const int gHH = (Hh * Hh) / 256;      // 2304
    const int gHD = (Hh * DFFf) / 256;    // 9216
    for (int l = 0; l < 2; ++l) {
        convw_k<<<gHH, 256, 0, stream>>>(wq + (size_t)l*Hh*Hh,   wtq[l], Hh,  Hh);
        convw_k<<<gHH, 256, 0, stream>>>(wk + (size_t)l*Hh*Hh,   wtk[l], Hh,  Hh);
        convw_k<<<gHH, 256, 0, stream>>>(wv + (size_t)l*Hh*Hh,   wtv[l], Hh,  Hh);
        convw_k<<<gHH, 256, 0, stream>>>(wo + (size_t)l*Hh*Hh,   wto[l], Hh,  Hh);
        convw_k<<<gHD, 256, 0, stream>>>(w1 + (size_t)l*Hh*DFFf, wt1[l], Hh,  DFFf);
        convw_k<<<gHD, 256, 0, stream>>>(w2 + (size_t)l*Hh*DFFf, wt2[l], DFFf, Hh);
    }

    dim3 gHHg(BT/256, Hh/64);    // (16,12)
    dim3 gHDg(BT/256, DFFf/64);  // (16,48)
    for (int l = 0; l < 2; ++l) {
        add2_k<<<(int)(nBTH/1024), 256, 0, stream>>>(h_cur, l == 0 ? in_h : h_cur, embs);
        ln_bf16_k<<<BT, 256, 0, stream>>>(h_cur, ln1_s + l*Hh, ln1_b + l*Hh, hn);
        gemm_bf16_k<<<gHHg, 256, 0, stream>>>(hn, wtq[l], bq + l*Hh, qb,  Hh, Hh, 1);
        gemm_bf16_k<<<gHHg, 256, 0, stream>>>(hn, wtk[l], bk + l*Hh, kb,  Hh, Hh, 1);
        gemm_bf16_k<<<gHHg, 256, 0, stream>>>(hn, wtv[l], bv + l*Hh, vtb, Hh, Hh, 2);
        attn_k<<<dim3(Bb*NHh, Tt/64), 128, 0, stream>>>(qb, kb, vtb, mask, avb);
        gemm_bf16_k<<<gHHg, 256, 0, stream>>>(avb, wto[l], bo + l*Hh, h_cur, Hh, Hh, 3);
        ln_bf16_k<<<BT, 256, 0, stream>>>(h_cur, ln2_s + l*Hh, ln2_b + l*Hh, hn);
        gemm_bf16_k<<<gHDg, 256, 0, stream>>>(hn,  wt1[l], b1 + l*DFFf, ffn,   DFFf, Hh,   4);
        gemm_bf16_k<<<gHHg, 256, 0, stream>>>(ffn, wt2[l], b2 + l*Hh,   h_cur, Hh,   DFFf, 3);
    }
    ln_f32_k<<<BT, 256, 0, stream>>>(h_cur, lnf_s, lnf_b, (float*)d_out);
}
